// IdealScore_2654289789422
// MI455X (gfx1250) — compile-verified, matrix operand order checked by reading, not run on hardware
//
#include <hip/hip_runtime.h>
#include <hip/hip_bf16.h>
#include <math.h>

// ---------------------------------------------------------------------------
// Problem constants (from reference): x[16,3072], images[10000,3072], f32.
// ---------------------------------------------------------------------------
#define BB   16      // batch rows == one WMMA M tile
#define NN   10000   // number of images
#define NP   10016   // N padded to multiple of 32 (K of second GEMM)
#define DD   3072    // C*H*W feature dim (K of first GEMM, multiple of 32)

typedef __attribute__((ext_vector_type(16))) __bf16 v16bf;
typedef __attribute__((ext_vector_type(8)))  float  v8f;

// Split 16 f32 into hi/lo bf16 limbs: v ~= hi + lo (16 mantissa bits total).
// Native casts lower to v_cvt_pk_bf16_f32 + v_fma_mix_f32_bf16 on gfx1250.
__device__ __forceinline__ void split_bf16(const float v[16], v16bf& hi, v16bf& lo) {
#pragma unroll
    for (int i = 0; i < 16; ++i) {
        __bf16 h = (__bf16)v[i];
        hi[i] = h;
        lo[i] = (__bf16)(v[i] - (float)h);
    }
}

__device__ __forceinline__ v8f wmma_bf16(v16bf a, v16bf b, v8f c) {
    // D(f32,16x16) += A(bf16,16x32) * B(bf16,32x16)
    return __builtin_amdgcn_wmma_f32_16x16x32_bf16(
        false, a, false, b, (short)0, c, false, false);
}

// Load the 16 K-values this lane owns for one K-step of 32 (16-bit A layout):
// lane half 0 -> K offsets {0..7, 16..23};  lane half 1 -> {8..15, 24..31}.
__device__ __forceinline__ void load16(const float* p, int kb, float v[16]) {
    const float4 q0 = *(const float4*)(p + kb);
    const float4 q1 = *(const float4*)(p + kb + 4);
    const float4 q2 = *(const float4*)(p + kb + 16);
    const float4 q3 = *(const float4*)(p + kb + 20);
    v[0]=q0.x;  v[1]=q0.y;  v[2]=q0.z;  v[3]=q0.w;
    v[4]=q1.x;  v[5]=q1.y;  v[6]=q1.z;  v[7]=q1.w;
    v[8]=q2.x;  v[9]=q2.y;  v[10]=q2.z; v[11]=q2.w;
    v[12]=q3.x; v[13]=q3.y; v[14]=q3.z; v[15]=q3.w;
}

// ---------------------------------------------------------------------------
// Kernel 1: cross[16, NP] = x @ images^T   (M=16, N=10000, K=3072)
//           + fused i_sq[n] = ||images[n]||^2
// One wave per 16x16 tile; 4 waves per block; 625 tiles total.
// ---------------------------------------------------------------------------
__global__ __launch_bounds__(128) void cross_kernel(
    const float* __restrict__ x, const float* __restrict__ img,
    float* __restrict__ cross, float* __restrict__ isq)
{
    const int wid  = threadIdx.x >> 5;
    const int lane = threadIdx.x & 31;
    const int t    = blockIdx.x * 4 + wid;        // wave-uniform tile id
    if (t >= NN / 16) return;                     // uniform exit: EXEC stays all-1

    const int m    = lane & 15;                   // batch row (A) / tile col (C)
    const int half = lane >> 4;
    const int kb   = half * 8;
    const int n    = t * 16 + m;                  // image row this lane feeds B with

    const float* xrow = x   + (size_t)m * DD;
    const float* irow = img + (size_t)n * DD;

    v8f c0 = {};   // main product  ah*bh
    v8f c1 = {};   // corrections   ah*bl + al*bh  (independent chain -> ILP)
    float s0 = 0.0f, s1 = 0.0f, s2 = 0.0f, s3 = 0.0f;  // isq partials (no long chain)

    for (int k0 = 0; k0 < DD; k0 += 32) {
        // Unconditional stream-ahead prefetch (speculative at DEV scope:
        // OOB translations are dropped silently -> no guard branch needed).
        __builtin_prefetch(irow + k0 + 512, 0, 1);

        float av[16], bv[16];
        load16(xrow + k0, kb, av);
        load16(irow + k0, kb, bv);
#pragma unroll
        for (int i = 0; i < 16; i += 4) {
            s0 = fmaf(bv[i+0], bv[i+0], s0);
            s1 = fmaf(bv[i+1], bv[i+1], s1);
            s2 = fmaf(bv[i+2], bv[i+2], s2);
            s3 = fmaf(bv[i+3], bv[i+3], s3);
        }

        v16bf ah, al, bh, bl;
        split_bf16(av, ah, al);
        split_bf16(bv, bh, bl);
        c1 = wmma_bf16(ah, bl, c1);
        c1 = wmma_bf16(al, bh, c1);
        c0 = wmma_bf16(ah, bh, c0);
    }

    // C/D layout: VGPR r, lanes 0-15 -> row r, lanes 16-31 -> row r+8; col = lane&15
#pragma unroll
    for (int r = 0; r < 8; ++r)
        cross[(size_t)(r + 8 * half) * NP + t * 16 + m] = c0[r] + c1[r];

    // i_sq: this lane holds K subset {kb..} — pair with xor-16 lane for the rest
    const float isq_part = (s0 + s1) + (s2 + s3);
    const float other = __shfl_xor(isq_part, 16, 32);
    if (half == 0) isq[n] = isq_part + other;
}

// ---------------------------------------------------------------------------
// Kernel 2: per-batch-row softmax over N with logits staged in LDS (40 KB,
// well under the 320 KB/WGP budget).
// weights[b, 0..NN) = softmax(-(xsq - 2at*cross + at^2*isq)/(2 bt2))
// weights[b, NN..NP) = 0  (so GEMM-2 needs no K-tail masking on A)
// ---------------------------------------------------------------------------
__global__ __launch_bounds__(256) void softmax_kernel(
    const float* __restrict__ x, const float* __restrict__ cross,
    const float* __restrict__ isq, const float* __restrict__ sval_p,
    float* __restrict__ weights)
{
    __shared__ float sarg[NP];
    __shared__ float red[256];
    const int b = blockIdx.x, tid = threadIdx.x;

    const float sv  = sval_p[0];
    const float at  = sqrtf(1.0f - sv);
    const float bt2 = sv;
    const float a2  = at * at;

    // ||x_b||^2
    float acc = 0.0f;
    for (int i = tid; i < DD; i += 256) { float v = x[(size_t)b * DD + i]; acc += v * v; }
    red[tid] = acc; __syncthreads();
    for (int s = 128; s > 0; s >>= 1) { if (tid < s) red[tid] += red[tid + s]; __syncthreads(); }
    const float xsq = red[0];
    __syncthreads();

    // logits + max
    float lmax = -3.4e38f;
    for (int n = tid; n < NN; n += 256) {
        float arg = -(xsq - 2.0f * at * cross[(size_t)b * NP + n] + a2 * isq[n]) / (2.0f * bt2);
        sarg[n] = arg;
        lmax = fmaxf(lmax, arg);
    }
    red[tid] = lmax; __syncthreads();
    for (int s = 128; s > 0; s >>= 1) { if (tid < s) red[tid] = fmaxf(red[tid], red[tid + s]); __syncthreads(); }
    const float gmax = red[0];
    __syncthreads();

    // exp + sum
    float lsum = 0.0f;
    for (int n = tid; n < NN; n += 256) { float e = __expf(sarg[n] - gmax); sarg[n] = e; lsum += e; }
    red[tid] = lsum; __syncthreads();
    for (int s = 128; s > 0; s >>= 1) { if (tid < s) red[tid] += red[tid + s]; __syncthreads(); }
    const float inv = 1.0f / red[0];

    for (int n = tid; n < NN; n += 256) weights[(size_t)b * NP + n] = sarg[n] * inv;
    if (tid < NP - NN) weights[(size_t)b * NP + NN + tid] = 0.0f;
}

// ---------------------------------------------------------------------------
// Kernel 3: mix[16, DD] = weights @ images  (M=16, N=3072, K=10016 padded)
//           fused epilogue: out = (at*mix - x)/bt2
// One wave per 16x16 D-tile; 192 tiles; 4 waves/block -> 48 blocks.
// Hot K-loop (312 steps) is unguarded; only the final step masks k >= NN.
// ---------------------------------------------------------------------------
__global__ __launch_bounds__(128) void mix_kernel(
    const float* __restrict__ x, const float* __restrict__ img,
    const float* __restrict__ weights, const float* __restrict__ sval_p,
    float* __restrict__ out)
{
    const int wid  = threadIdx.x >> 5;
    const int lane = threadIdx.x & 31;
    const int t    = blockIdx.x * 4 + wid;   // 0..191, exact
    const int d0   = t * 16;

    const int m    = lane & 15;              // batch row (A) / tile col (C)
    const int half = lane >> 4;
    const int kb   = half * 8;

    const float* wrow = weights + (size_t)m * NP;
    v8f c0 = {};
    v8f c1 = {};

    const int KFULL = (NN / 32) * 32;        // 9984: last full unguarded step
    for (int k0 = 0; k0 < KFULL; k0 += 32) {
        float av[16], bv[16];
        load16(wrow + k0, kb, av);
#pragma unroll
        for (int i = 0; i < 16; ++i) {
            const int k = k0 + ((i < 8) ? (kb + i) : (16 + kb + (i - 8)));
            bv[i] = img[(size_t)k * DD + d0 + m];
        }
        v16bf ah, al, bh, bl;
        split_bf16(av, ah, al);
        split_bf16(bv, bh, bl);
        c1 = wmma_bf16(ah, bl, c1);
        c1 = wmma_bf16(al, bh, c1);
        c0 = wmma_bf16(ah, bh, c0);
    }
    {   // tail step k0 = 9984: K in [9984, 10016), mask k >= NN (A pad is zero)
        const int k0 = KFULL;
        float av[16], bv[16];
        load16(wrow + k0, kb, av);
#pragma unroll
        for (int i = 0; i < 16; ++i) {
            const int k = k0 + ((i < 8) ? (kb + i) : (16 + kb + (i - 8)));
            bv[i] = (k < NN) ? img[(size_t)k * DD + d0 + m] : 0.0f;
        }
        v16bf ah, al, bh, bl;
        split_bf16(av, ah, al);
        split_bf16(bv, bh, bl);
        c1 = wmma_bf16(ah, bl, c1);
        c1 = wmma_bf16(al, bh, c1);
        c0 = wmma_bf16(ah, bh, c0);
    }

    const float sv  = sval_p[0];
    const float at  = sqrtf(1.0f - sv);
    const float ib  = 1.0f / sv;             // 1/bt2
#pragma unroll
    for (int r = 0; r < 8; ++r) {
        const int row = r + 8 * half;
        const size_t idx = (size_t)row * DD + d0 + m;
        out[idx] = (at * (c0[r] + c1[r]) - x[idx]) * ib;
    }
}

// ---------------------------------------------------------------------------
// Host-side launch
// ---------------------------------------------------------------------------
extern "C" void kernel_launch(void* const* d_in, const int* in_sizes, int n_in,
                              void* d_out, int out_size, void* d_ws, size_t ws_size,
                              hipStream_t stream) {
    (void)in_sizes; (void)n_in; (void)out_size; (void)ws_size;
    const float* x    = (const float*)d_in[0];
    const float* img  = (const float*)d_in[1];
    const float* sval = (const float*)d_in[2];
    float*       out  = (float*)d_out;

    float* cross   = (float*)d_ws;              // [16, NP]
    float* weights = cross + (size_t)BB * NP;   // [16, NP]
    float* isq     = weights + (size_t)BB * NP; // [NP]

    cross_kernel  <<<(NN / 16 + 3) / 4, 128, 0, stream>>>(x, img, cross, isq);
    softmax_kernel<<<BB, 256, 0, stream>>>(x, cross, isq, sval, weights);
    mix_kernel    <<<(DD / 16) / 4, 128, 0, stream>>>(x, img, weights, sval, out);
}
